// GNN_5540507812348
// MI455X (gfx1250) — matrix operand order repaired and checked by reading, not compiled, not run on hardware
//
#include <hip/hip_runtime.h>
#include <hip/hip_bf16.h>

typedef _Float16 half_t;
typedef __attribute__((ext_vector_type(16))) _Float16 v16h;
typedef __attribute__((ext_vector_type(8)))  float    v8f;

#define D_NODE 10
#define D_IN   9
#define D_EDGE 11
#define D_OUT  9
#define H_MSG  96
#define GRU_IN 20
#define N_ITERS 7

#define W1_FRAGS 6    // 6 N-tiles, 1 kstep (K=32 covers 20)
#define W2_FRAGS 18   // tn*3 + ks
#define W3_FRAGS 3    // 3 ksteps, 1 N-tile (N=16 covers 11)
#define FRAG_HALVES 512   // 32 lanes * 16 halves per B fragment
#define WAVES 8

// ---------------------------------------------------------------------------
// Pack W1/W2/W3 (f32, row-major (out,in)) into per-lane-ordered f16 B-matrix
// fragments for v_wmma_f32_16x16x32_f16.
// B layout (32x16, KxN): lane l -> N = l%16, half-group hg = l/16;
// element j (0..15) -> K = (j<8 ? j : j+8) + hg*8. Stored frag-major, lane-major.
// ---------------------------------------------------------------------------
__global__ void gnn_prep_wfrag(const float* __restrict__ W1,
                               const float* __restrict__ W2,
                               const float* __restrict__ W3,
                               half_t* __restrict__ wfrag) {
  const int total = (W1_FRAGS + W2_FRAGS + W3_FRAGS) * FRAG_HALVES;
  for (int idx = threadIdx.x + blockIdx.x * blockDim.x; idx < total;
       idx += blockDim.x * gridDim.x) {
    int f = idx / FRAG_HALVES;
    int r = idx % FRAG_HALVES;
    int lane = r >> 4;
    int j = r & 15;
    int n = lane & 15;
    int hg = lane >> 4;
    int K = ((j < 8) ? j : (j + 8)) + hg * 8;  // K within one 32-deep kstep
    float v = 0.f;
    if (f < W1_FRAGS) {
      int N = f * 16 + n;                      // 0..95
      if (K < GRU_IN) v = W1[N * GRU_IN + K];  // W1: (96,20)
    } else if (f < W1_FRAGS + W2_FRAGS) {
      int f2 = f - W1_FRAGS;
      int tn = f2 / 3, ks = f2 % 3;
      int N = tn * 16 + n;
      int Kg = ks * 32 + K;                    // 0..95
      v = W2[N * H_MSG + Kg];                  // W2: (96,96)
    } else {
      int ks = f - W1_FRAGS - W2_FRAGS;
      int Kg = ks * 32 + K;
      if (n < D_EDGE) v = W3[n * H_MSG + Kg];  // W3: (11,96)
    }
    wfrag[idx] = (half_t)v;
  }
}

// ---------------------------------------------------------------------------
// Fused edge MLP: stage h[src]||h[dst] through LDS -> (16x32) f16 A tile per
// wave, run 20->96->96->11 with v_wmma_f32_16x16x32_f16 (f32 accum), then
// scatter-add messages into agg via global_atomic_add_f32.
// A layout (16x32 f16): lane l -> M = l%16; element j -> K = (j<8? j : j+8) + (l/16)*8.
// D layout (16x16 f32): lane l -> N = l%16; VGPR r -> M = r + 8*(l/16).
// ---------------------------------------------------------------------------
__global__ __launch_bounds__(256)
void gnn_edge_mlp(const float* __restrict__ hstate,
                  const int* __restrict__ src_ids,
                  const int* __restrict__ dst_ids,
                  const half_t* __restrict__ wfrag,
                  const float* __restrict__ b1,
                  const float* __restrict__ b2,
                  const float* __restrict__ b3,
                  float* __restrict__ agg,
                  int n_edges) {
  __shared__ half_t act[WAVES][2][16][H_MSG];  // 48 KB: per-wave layer tiles
  __shared__ half_t inbuf[WAVES][16][32];      //  8 KB: staged [h_src||h_dst|0]
  __shared__ int    dstbuf[WAVES][16];         // dst ids of the current tile

  const int wave = threadIdx.x >> 5;
  const int lane = threadIdx.x & 31;
  const int hg = lane >> 4;     // lane half-group
  const int m  = lane & 15;     // A-row index / D-column index

  const half_t* fragW1 = wfrag;
  const half_t* fragW2 = wfrag + W1_FRAGS * FRAG_HALVES;
  const half_t* fragW3 = wfrag + (W1_FRAGS + W2_FRAGS) * FRAG_HALVES;

  // Zero the K-pad columns (20..31) once; never rewritten inside the loop.
  for (int t = lane; t < 16 * 12; t += 32) {
    inbuf[wave][t / 12][20 + t % 12] = (half_t)0.f;
  }
  asm volatile("s_wait_dscnt 0" ::: "memory");

  const int n_tiles = (n_edges + 15) >> 4;
  for (int tile = blockIdx.x * WAVES + wave; tile < n_tiles;
       tile += gridDim.x * WAVES) {
    // ----- stage inputs: group 0 loads src row, group 1 loads dst row -----
    int e  = tile * 16 + m;
    int eL = e < n_edges ? e : (n_edges - 1);
    int node = (hg == 0) ? src_ids[eL] : dst_ids[eL];
    if (hg == 1) dstbuf[wave][m] = node;
    const float* rowp = hstate + (size_t)node * D_NODE;
    const int cbase = hg * D_NODE;
    #pragma unroll
    for (int k = 0; k < D_NODE; k += 2) {
      float2 v = *(const float2*)(rowp + k);   // h rows are 8B aligned
      inbuf[wave][m][cbase + k]     = (half_t)v.x;
      inbuf[wave][m][cbase + k + 1] = (half_t)v.y;
    }
    asm volatile("s_wait_dscnt 0" ::: "memory");

    // ----- layer 1: (16x32) x (32x96) -----
    v16h a1;
    #pragma unroll
    for (int j = 0; j < 16; ++j) {
      int K = ((j < 8) ? j : (j + 8)) + hg * 8;
      a1[j] = inbuf[wave][m][K];
    }
    #pragma unroll
    for (int tn = 0; tn < 6; ++tn) {
      v16h b = *(const v16h*)(fragW1 + tn * FRAG_HALVES + lane * 16);
      v8f acc = {};
      acc = __builtin_amdgcn_wmma_f32_16x16x32_f16(false, a1, false, b,
                                                   (short)0, acc, false, false);
      int o = tn * 16 + m;           // output feature (D column)
      float bias = b1[o];
      #pragma unroll
      for (int r = 0; r < 8; ++r) {
        int row = r + 8 * hg;        // edge row (D row)
        float x = acc[r] + bias;
        act[wave][0][row][o] = (half_t)(x > 0.f ? x : 0.f);
      }
    }
    asm volatile("s_wait_dscnt 0" ::: "memory");

    // ----- layer 2: 96 -> 96 -----
    #pragma unroll
    for (int tn = 0; tn < 6; ++tn) {
      v8f acc = {};
      #pragma unroll
      for (int ks = 0; ks < 3; ++ks) {
        v16h a;
        #pragma unroll
        for (int j = 0; j < 16; ++j) {
          int K = ks * 32 + ((j < 8) ? j : (j + 8)) + hg * 8;
          a[j] = act[wave][0][m][K];
        }
        v16h b = *(const v16h*)(fragW2 + (tn * 3 + ks) * FRAG_HALVES + lane * 16);
        acc = __builtin_amdgcn_wmma_f32_16x16x32_f16(false, a, false, b,
                                                     (short)0, acc, false, false);
      }
      int o = tn * 16 + m;
      float bias = b2[o];
      #pragma unroll
      for (int r = 0; r < 8; ++r) {
        int row = r + 8 * hg;
        float x = acc[r] + bias;
        act[wave][1][row][o] = (half_t)(x > 0.f ? x : 0.f);
      }
    }
    asm volatile("s_wait_dscnt 0" ::: "memory");

    // ----- layer 3: 96 -> 11, then scatter-add into agg[dst] -----
    v8f acc3 = {};
    #pragma unroll
    for (int ks = 0; ks < 3; ++ks) {
      v16h a;
      #pragma unroll
      for (int j = 0; j < 16; ++j) {
        int K = ks * 32 + ((j < 8) ? j : (j + 8)) + hg * 8;
        a[j] = act[wave][1][m][K];
      }
      v16h b = *(const v16h*)(fragW3 + ks * FRAG_HALVES + lane * 16);
      acc3 = __builtin_amdgcn_wmma_f32_16x16x32_f16(false, a, false, b,
                                                    (short)0, acc3, false, false);
    }
    if (m < D_EDGE) {                // D column = message feature, 11 valid
      float bias = b3[m];
      #pragma unroll
      for (int r = 0; r < 8; ++r) {
        int row = r + 8 * hg;
        int ee = tile * 16 + row;
        if (ee < n_edges) {
          int dd = dstbuf[wave][row];
          atomicAdd(&agg[dd * D_EDGE + m], acc3[r] + bias);
        }
      }
    }
  }
}

// ---------------------------------------------------------------------------
// Per-node GRU + output head. softmax(log_softmax(x)) == softmax(x), so the
// head reduces to a plain softmax of the logits. Also re-zeroes agg for the
// next iteration after consuming it.
// ---------------------------------------------------------------------------
__global__ __launch_bounds__(256)
void gnn_node_update(const float* __restrict__ node_inputs,
                     float* __restrict__ agg,
                     const float* __restrict__ Wih,
                     const float* __restrict__ Whh,
                     const float* __restrict__ bih,
                     const float* __restrict__ bhh,
                     const float* __restrict__ Wo,
                     const float* __restrict__ bo,
                     float* __restrict__ hstate,
                     float* __restrict__ out_it,
                     int n_nodes) {
  __shared__ float sWih[3 * D_NODE * GRU_IN];   // 600
  __shared__ float sWhh[3 * D_NODE * D_NODE];   // 300
  __shared__ float sWo[D_OUT * D_NODE];         // 90
  __shared__ float sbih[3 * D_NODE], sbhh[3 * D_NODE], sbo[D_OUT];
  for (int i = threadIdx.x; i < 3 * D_NODE * GRU_IN; i += blockDim.x) sWih[i] = Wih[i];
  for (int i = threadIdx.x; i < 3 * D_NODE * D_NODE; i += blockDim.x) sWhh[i] = Whh[i];
  for (int i = threadIdx.x; i < D_OUT * D_NODE;      i += blockDim.x) sWo[i]  = Wo[i];
  if (threadIdx.x < 3 * D_NODE) { sbih[threadIdx.x] = bih[threadIdx.x];
                                  sbhh[threadIdx.x] = bhh[threadIdx.x]; }
  if (threadIdx.x < D_OUT) sbo[threadIdx.x] = bo[threadIdx.x];
  __syncthreads();

  int nidx = blockIdx.x * blockDim.x + threadIdx.x;
  if (nidx >= n_nodes) return;

  float x[GRU_IN];
  #pragma unroll
  for (int k = 0; k < D_EDGE; ++k) {
    x[k] = agg[nidx * D_EDGE + k];
    agg[nidx * D_EDGE + k] = 0.f;   // ready for next iteration's scatter
  }
  #pragma unroll
  for (int k = 0; k < D_IN; ++k) x[D_EDGE + k] = node_inputs[nidx * D_IN + k];
  float hc[D_NODE];
  #pragma unroll
  for (int k = 0; k < D_NODE; ++k) hc[k] = hstate[nidx * D_NODE + k];

  float gi[3 * D_NODE], gh[3 * D_NODE];
  #pragma unroll
  for (int o = 0; o < 3 * D_NODE; ++o) {
    float s1 = sbih[o];
    #pragma unroll
    for (int k = 0; k < GRU_IN; ++k) s1 += sWih[o * GRU_IN + k] * x[k];
    gi[o] = s1;
    float s2 = sbhh[o];
    #pragma unroll
    for (int k = 0; k < D_NODE; ++k) s2 += sWhh[o * D_NODE + k] * hc[k];
    gh[o] = s2;
  }
  float hn[D_NODE];
  #pragma unroll
  for (int i = 0; i < D_NODE; ++i) {
    float r  = 1.f / (1.f + __expf(-(gi[i] + gh[i])));
    float z  = 1.f / (1.f + __expf(-(gi[D_NODE + i] + gh[D_NODE + i])));
    float nn = tanhf(gi[2 * D_NODE + i] + r * gh[2 * D_NODE + i]);
    hn[i] = (1.f - z) * nn + z * hc[i];
  }
  #pragma unroll
  for (int i = 0; i < D_NODE; ++i) hstate[nidx * D_NODE + i] = hn[i];

  float logit[D_OUT];
  float mx = -1e30f;
  #pragma unroll
  for (int o = 0; o < D_OUT; ++o) {
    float s = sbo[o];
    #pragma unroll
    for (int k = 0; k < D_NODE; ++k) s += sWo[o * D_NODE + k] * hn[k];
    logit[o] = s;
    mx = fmaxf(mx, s);
  }
  float ssum = 0.f;
  #pragma unroll
  for (int o = 0; o < D_OUT; ++o) { logit[o] = __expf(logit[o] - mx); ssum += logit[o]; }
  float inv = 1.f / ssum;
  #pragma unroll
  for (int o = 0; o < D_OUT; ++o) out_it[nidx * D_OUT + o] = logit[o] * inv;
}

// ---------------------------------------------------------------------------
static inline size_t align256(size_t x) { return (x + 255) & ~(size_t)255; }

extern "C" void kernel_launch(void* const* d_in, const int* in_sizes, int n_in,
                              void* d_out, int out_size, void* d_ws, size_t ws_size,
                              hipStream_t stream) {
  const float* node_inputs = (const float*)d_in[0];
  const int*   src_ids     = (const int*)d_in[1];
  const int*   dst_ids     = (const int*)d_in[2];
  const float* W1  = (const float*)d_in[3];
  const float* b1  = (const float*)d_in[4];
  const float* W2  = (const float*)d_in[5];
  const float* b2  = (const float*)d_in[6];
  const float* W3  = (const float*)d_in[7];
  const float* b3  = (const float*)d_in[8];
  const float* Wo  = (const float*)d_in[9];
  const float* bo  = (const float*)d_in[10];
  const float* Wih = (const float*)d_in[11];
  const float* Whh = (const float*)d_in[12];
  const float* bih = (const float*)d_in[13];
  const float* bhh = (const float*)d_in[14];

  const int n_nodes = in_sizes[0] / D_IN;
  const int n_edges = in_sizes[1];

  char* ws = (char*)d_ws;
  size_t h_bytes   = align256((size_t)n_nodes * D_NODE * sizeof(float));
  size_t agg_bytes = align256((size_t)n_nodes * D_EDGE * sizeof(float));
  float*  hstate = (float*)ws;
  float*  agg    = (float*)(ws + h_bytes);
  half_t* wfrag  = (half_t*)(ws + h_bytes + agg_bytes);

  hipMemsetAsync(hstate, 0, (size_t)n_nodes * D_NODE * sizeof(float), stream);
  hipMemsetAsync(agg,    0, (size_t)n_nodes * D_EDGE * sizeof(float), stream);

  gnn_prep_wfrag<<<8, 256, 0, stream>>>(W1, W2, W3, wfrag);

  const int n_tiles = (n_edges + 15) >> 4;
  int edge_blocks = (n_tiles + WAVES - 1) / WAVES;
  if (edge_blocks > 8192) edge_blocks = 8192;   // grid-stride handles the rest
  int node_blocks = (n_nodes + 255) / 256;

  for (int it = 0; it < N_ITERS; ++it) {
    gnn_edge_mlp<<<edge_blocks, 256, 0, stream>>>(
        hstate, src_ids, dst_ids, wfrag, b1, b2, b3, agg, n_edges);
    gnn_node_update<<<node_blocks, 256, 0, stream>>>(
        node_inputs, agg, Wih, Whh, bih, bhh, Wo, bo, hstate,
        (float*)d_out + (size_t)it * n_nodes * D_OUT, n_nodes);
  }
}
